// MultiHeadAttention_28363964022929
// MI455X (gfx1250) — matrix operand order repaired and checked by reading, not compiled, not run on hardware
//
#include <hip/hip_runtime.h>

typedef __attribute__((ext_vector_type(16))) _Float16 v16h;
typedef __attribute__((ext_vector_type(8)))  float    v8f;

#define N_HEADS   8
#define INPUT_DIM 128
#define EMBED_DIM 128
#define KEY_DIM   16
#define BATCH     4
#define GRAPH     2048
#define SCALE     0.25f   // 1/sqrt(16)

static __device__ __forceinline__ v8f wmma_f16(v16h a, v16h b, v8f c) {
  // D = A(16x32 f16) * B(32x16 f16) + C(16x16 f32)
  return __builtin_amdgcn_wmma_f32_16x16x32_f16(
      /*neg_a=*/false, a, /*neg_b=*/false, b,
      /*c_mod=*/(short)0, c, /*reuse_a=*/false, /*reuse_b=*/false);
}

// ---------------------------------------------------------------------------
// Kernel 1: per-head QKV projection.  One wave computes a 16x16 tile of
// Q, K and V for one (head, batch, row-tile), reusing the h A-fragments
// across the three weight matrices (12 WMMAs / wave, K-loop 128 = 4x32).
// Q and K are stored row-major [H,B,N,dk]; V is stored TRANSPOSED
// [H,B,dv,N] so the flash kernel's V B-fragments are contiguous per lane.
// ---------------------------------------------------------------------------
__global__ void qkv_project_kernel(const float* __restrict__ hsrc,
                                   const float* __restrict__ Wq,
                                   const float* __restrict__ Wk,
                                   const float* __restrict__ Wv,
                                   _Float16* __restrict__ Qh,
                                   _Float16* __restrict__ Kh,
                                   _Float16* __restrict__ Vt) {
  const int lane = threadIdx.x & 31;
  const int wid  = (blockIdx.x * blockDim.x + threadIdx.x) >> 5;
  const int qt   = wid & (GRAPH / 16 - 1);        // 0..127 row tile
  const int b    = (wid >> 7) & (BATCH - 1);
  const int head = wid >> 9;
  const int n15  = lane & 15;
  const int hiA  = (lane & 16) ? 8 : 0;           // A: hi half-wave holds K+8
  const int hiB  = (lane & 16) ? 16 : 0;          // unused; B also takes hiA (hi half-wave holds K+8)
  const int m0   = qt * 16;

  // A fragments: rows of h (f32 -> f16).  lane<16: K {0..7,16..23}; +8 for hi.
  v16h a[4];
  const float* hrow = hsrc + ((size_t)b * GRAPH + m0 + n15) * INPUT_DIM;
#pragma unroll
  for (int s = 0; s < 4; ++s) {
#pragma unroll
    for (int e = 0; e < 16; ++e) {
      const int kl = (e < 8 ? e : e + 8) + hiA;
      a[s][e] = (_Float16)hrow[s * 32 + kl];
    }
  }

  const float* Ws[3] = { Wq, Wk, Wv };
#pragma unroll
  for (int w = 0; w < 3; ++w) {
    const float* W = Ws[w] + (size_t)head * INPUT_DIM * KEY_DIM;
    v8f c = {};
#pragma unroll
    for (int s = 0; s < 4; ++s) {
      v16h bf;   // B: column n = lane&15, K {0..7,16..23} (+8 for hi half-wave)
#pragma unroll
      for (int e = 0; e < 16; ++e) {
        const int kl = s * 32 + (e < 8 ? e : e + 8) + hiA;
        bf[e] = (_Float16)W[(size_t)kl * KEY_DIM + n15];
      }
      c = wmma_f16(a[s], bf, c);
    }
    if (w < 2) {
      _Float16* out = (w == 0 ? Qh : Kh) +
                      (((size_t)head * BATCH + b) * GRAPH + m0) * KEY_DIM;
#pragma unroll
      for (int r = 0; r < 8; ++r) {
        const int rr = r + ((lane & 16) ? 8 : 0);   // C/D layout: row rr, col n15
        out[(size_t)rr * KEY_DIM + n15] = (_Float16)c[r];
      }
    } else {
      // transposed store: Vt[((h*B+b)*dv + col)*GRAPH + row]
      _Float16* out = Vt + (((size_t)head * BATCH + b) * KEY_DIM + n15) * GRAPH + m0;
#pragma unroll
      for (int r = 0; r < 8; ++r) {
        const int rr = r + ((lane & 16) ? 8 : 0);
        out[rr] = (_Float16)c[r];
      }
    }
  }
}

// ---------------------------------------------------------------------------
// Kernel 2: flash attention, 32 keys per iteration.
//   S0,S1 = Q(16x16, dk padded to 32) x K^T            (2 padded WMMAs)
//   O    += P(16x32, full contraction) x V(32x16)      (1 full WMMA)
// Online softmax state per row lives in the 16-lane half-wave that owns the
// row (C/D layout); P converts C/D->A layout through a 1KB/wave LDS tile.
// ---------------------------------------------------------------------------
__global__ void flash_attn_kernel(const _Float16* __restrict__ Qh,
                                  const _Float16* __restrict__ Kh,
                                  const _Float16* __restrict__ Vt,
                                  _Float16* __restrict__ Hh) {
  __shared__ _Float16 plds[8][16 * 32];           // per-wave P tile (16q x 32k)
  const int  lane   = threadIdx.x & 31;
  const int  wslot  = threadIdx.x >> 5;
  const int  wid    = (blockIdx.x * blockDim.x + threadIdx.x) >> 5;
  const int  qt     = wid & (GRAPH / 16 - 1);
  const int  b      = (wid >> 7) & (BATCH - 1);
  const int  head   = wid >> 9;
  const int  n15    = lane & 15;
  const bool hihalf = (lane & 16) != 0;

  const size_t hb = ((size_t)head * BATCH + b) * GRAPH;
  // per-lane V column base (column = dv index n15; transposed layout)
  const _Float16* Vcol = Vt + (((size_t)head * BATCH + b) * KEY_DIM + n15) * GRAPH
                            + (hihalf ? 16 : 0);

  // Q A-fragment, contraction (dk) padded 16->32 with zeros.
  v16h aq;
  {
    const _Float16* qrow = Qh + (hb + qt * 16 + n15) * KEY_DIM;
#pragma unroll
    for (int e = 0; e < 16; ++e) {
      const int kl = e + (hihalf ? 16 : 0);
      aq[e] = (kl < KEY_DIM) ? qrow[kl] : (_Float16)0.0f;
    }
  }

  float m_i[8], l_i[8];
  v8f o = {};
#pragma unroll
  for (int r = 0; r < 8; ++r) { m_i[r] = -1e30f; l_i[r] = 0.0f; }

  for (int kt = 0; kt < GRAPH / 32; ++kt) {
    const int k0 = kt * 32;
    if (kt + 1 < GRAPH / 32) {   // hide L2/HBM latency behind softmax VALU work
      __builtin_prefetch(Kh + (hb + k0 + 32 + n15) * KEY_DIM, 0, 0);
      __builtin_prefetch(Vcol + k0 + 32, 0, 0);
    }

    // K^T B-fragments for keys [k0,k0+16) and [k0+16,k0+32): column n = key,
    // contraction = dk (lanes 16-31 = zero pad).
    v16h bk0, bk1;
    const _Float16* kr0 = Kh + (hb + k0 + n15) * KEY_DIM;
#pragma unroll
    for (int e = 0; e < 16; ++e) {
      bk0[e] = hihalf ? (_Float16)0.0f : kr0[e];
      bk1[e] = hihalf ? (_Float16)0.0f : kr0[16 * KEY_DIM + e];
    }

    v8f z = {};
    v8f s0 = wmma_f16(aq, bk0, z);
    v8f s1 = wmma_f16(aq, bk1, z);

    // Online softmax over the 32 keys (row r of the half-wave in elem r).
#pragma unroll
    for (int r = 0; r < 8; ++r) {
      const float x0 = s0[r] * SCALE;
      const float x1 = s1[r] * SCALE;
      float rm = fmaxf(x0, x1);
      rm = fmaxf(rm, __shfl_xor(rm, 1, 32));
      rm = fmaxf(rm, __shfl_xor(rm, 2, 32));
      rm = fmaxf(rm, __shfl_xor(rm, 4, 32));
      rm = fmaxf(rm, __shfl_xor(rm, 8, 32));
      const float mnew = fmaxf(m_i[r], rm);
      const float corr = __expf(m_i[r] - mnew);
      const float p0   = __expf(x0 - mnew);
      const float p1   = __expf(x1 - mnew);
      float rs = p0 + p1;
      rs += __shfl_xor(rs, 1, 32);
      rs += __shfl_xor(rs, 2, 32);
      rs += __shfl_xor(rs, 4, 32);
      rs += __shfl_xor(rs, 8, 32);
      l_i[r] = l_i[r] * corr + rs;
      o[r]  *= corr;
      m_i[r] = mnew;
      const int rr = r + (hihalf ? 8 : 0);
      plds[wslot][rr * 32 + n15]      = (_Float16)p0;   // keys k0..k0+15
      plds[wslot][rr * 32 + 16 + n15] = (_Float16)p1;   // keys k0+16..k0+31
    }
    __builtin_amdgcn_wave_barrier();
    asm volatile("s_wait_dscnt 0" ::: "memory");  // LDS in-order per wave

    // P A-fragment: full 16x32 contraction over the 32 keys.
    v16h ap;
#pragma unroll
    for (int e = 0; e < 16; ++e) {
      const int kk = e + (hihalf ? 16 : 0);
      ap[e] = plds[wslot][n15 * 32 + kk];
    }

    // V B-fragment: 32 contiguous bytes per lane (transposed V layout).
    v16h bv;
    const _Float16* vp = Vcol + k0;
#pragma unroll
    for (int e = 0; e < 16; ++e)
      bv[e] = vp[e];

    o = wmma_f16(ap, bv, o);
    __builtin_amdgcn_wave_barrier();              // keep LDS WAR ordered
  }

  _Float16* out = Hh + (hb + qt * 16) * KEY_DIM;
#pragma unroll
  for (int r = 0; r < 8; ++r) {
    const int rr = r + (hihalf ? 8 : 0);
    out[(size_t)rr * KEY_DIM + n15] = (_Float16)(o[r] / l_i[r]);
  }
}

// ---------------------------------------------------------------------------
// Kernel 3: output projection.  heads concatenated over h == [B*N,128] and
// W_out's [H,16,128] layout flattens to the stacked [128,128] matrix, so this
// is a plain [8192,128]x[128,128] f16-WMMA GEMM with f32 output.
// ---------------------------------------------------------------------------
__global__ void out_proj_kernel(const _Float16* __restrict__ Hh,
                                const float* __restrict__ Wo,
                                float* __restrict__ out) {
  const int  lane   = threadIdx.x & 31;
  const int  wid    = (blockIdx.x * blockDim.x + threadIdx.x) >> 5;
  const int  mt     = wid & (BATCH * GRAPH / 16 - 1);  // 0..511 row tile
  const int  nt     = wid >> 9;                        // 0..7 col tile
  const int  n15    = lane & 15;
  const bool hihalf = (lane & 16) != 0;

  const int row0 = mt * 16;
  const int b    = row0 >> 11;            // GRAPH = 2048
  const int q0   = row0 & (GRAPH - 1);

  v8f c = {};
#pragma unroll
  for (int s = 0; s < 4; ++s) {
    v16h a, bf;
#pragma unroll
    for (int e = 0; e < 16; ++e) {
      const int kl = s * 32 + e + (hihalf ? 16 : 0);
      const int hd = kl >> 4, kk = kl & 15;
      a[e] = Hh[(((size_t)hd * BATCH + b) * GRAPH + q0 + n15) * KEY_DIM + kk];
    }
#pragma unroll
    for (int e = 0; e < 16; ++e) {
      const int kl = s * 32 + e + (hihalf ? 16 : 0);
      bf[e] = (_Float16)Wo[(size_t)kl * EMBED_DIM + nt * 16 + n15];
    }
    c = wmma_f16(a, bf, c);
  }
#pragma unroll
  for (int r = 0; r < 8; ++r) {
    const int rr = r + (hihalf ? 8 : 0);
    out[((size_t)b * GRAPH + q0 + rr) * EMBED_DIM + nt * 16 + n15] = c[r];
  }
}

// ---------------------------------------------------------------------------
extern "C" void kernel_launch(void* const* d_in, const int* in_sizes, int n_in,
                              void* d_out, int out_size, void* d_ws, size_t ws_size,
                              hipStream_t stream) {
  const float* h  = (const float*)d_in[0];
  const float* Wq = (const float*)d_in[1];
  const float* Wk = (const float*)d_in[2];
  const float* Wv = (const float*)d_in[3];
  const float* Wo = (const float*)d_in[4];
  float* out = (float*)d_out;

  // Workspace: 4 f16 buffers of H*B*N*16 = 1Mi elements (2 MB each) = 8 MB.
  const size_t elems = (size_t)N_HEADS * BATCH * GRAPH * KEY_DIM;
  _Float16* Qh = (_Float16*)d_ws;
  _Float16* Kh = Qh + elems;
  _Float16* Vt = Kh + elems;   // transposed: [H,B,dv,N]
  _Float16* Hh = Vt + elems;

  const int waves12 = N_HEADS * BATCH * (GRAPH / 16);            // 4096 waves
  const int waves3  = (BATCH * GRAPH / 16) * (EMBED_DIM / 16);   // 4096 waves

  qkv_project_kernel<<<waves12 / 8, 256, 0, stream>>>(h, Wq, Wk, Wv, Qh, Kh, Vt);
  flash_attn_kernel <<<waves12 / 8, 256, 0, stream>>>(Qh, Kh, Vt, Hh);
  out_proj_kernel   <<<waves3  / 8, 256, 0, stream>>>(Hh, Wo, out);
}
